// my_lstm_42262478193021
// MI455X (gfx1250) — matrix operand order repaired and checked by reading, not compiled, not run on hardware
//
#include <hip/hip_runtime.h>

typedef __attribute__((ext_vector_type(16))) __bf16 v16bf;
typedef __attribute__((ext_vector_type(8)))  float  v8f;

#define T_LEN 4096
#define HID   1024
#define G4    4096   // 4*HID
#define NWG   32     // persistent workgroups for recurrence
#define RBLK  256    // threads per block

// ---------------------------------------------------------------------------
// helpers
// ---------------------------------------------------------------------------
__device__ __forceinline__ unsigned short f2bf(float f) {
    unsigned u = __float_as_uint(f);
    unsigned r = u + 0x7FFFu + ((u >> 16) & 1u);   // round-to-nearest-even
    return (unsigned short)(r >> 16);
}

__device__ __forceinline__ float sigmoidf_(float x) {
    return 1.0f / (1.0f + __expf(-x));
}

// LDS byte offset of a generic pointer into __shared__ (low 32 bits of the
// shared-aperture generic address are the LDS offset on AMDGPU).
__device__ __forceinline__ unsigned lds_off(const void* p) {
    return (unsigned)(unsigned long long)p;
}

// CDNA5 async global->LDS copy, 16 bytes per active lane (ASYNCcnt tracked).
__device__ __forceinline__ void async_cp16(unsigned lds_byte_off, const void* gaddr) {
    asm volatile("global_load_async_to_lds_b128 %0, %1, off"
                 :: "v"(lds_byte_off), "v"(gaddr) : "memory");
}

__device__ __forceinline__ void wait_async0() {
#if __has_builtin(__builtin_amdgcn_s_wait_asynccnt)
    __builtin_amdgcn_s_wait_asynccnt(0);
#else
    asm volatile("s_wait_asynccnt 0x0" ::: "memory");
#endif
}

union FragU {
    v16bf v;
    uint4 q[2];
};

__device__ __forceinline__ v16bf ld_frag(const unsigned short* p0,
                                         const unsigned short* p1) {
    FragU u;
    u.q[0] = *reinterpret_cast<const uint4*>(p0);
    u.q[1] = *reinterpret_cast<const uint4*>(p1);
    return u.v;
}

__device__ __forceinline__ v8f wmma_bf16(v16bf a, v16bf b, v8f c) {
    return __builtin_amdgcn_wmma_f32_16x16x32_bf16(
        false, a, false, b, (short)0, c, false, false);
}

// ---------------------------------------------------------------------------
// fp32 -> bf16 conversion (grid-stride)
// ---------------------------------------------------------------------------
__global__ void k_f2bf(const float* __restrict__ in,
                       unsigned short* __restrict__ out, long long n) {
    long long i = (long long)blockIdx.x * blockDim.x + threadIdx.x;
    long long stride = (long long)gridDim.x * blockDim.x;
    for (; i < n; i += stride) out[i] = f2bf(in[i]);
}

// zero barrier counters (ws is poisoned; must re-zero every call)
__global__ void k_init(unsigned int* counters) {
    if (threadIdx.x < 64) counters[threadIdx.x] = 0u;
}

// ---------------------------------------------------------------------------
// Input-projection GEMM:  Gpre[z][t][j] = sum_k A[z][t][k]*W[j][k] + bih[j]+bhh[j]
// A: bf16 [2][T][K] row-major, W: bf16 [4096][K] row-major (K = 1024)
// Tile: 128(M) x 64(N), K-chunk 32, double-buffered async-to-LDS staging.
// 8 waves: wave = M-subtile, 4 N-subtiles each.
// ---------------------------------------------------------------------------
__global__ void k_gemm(const unsigned short* __restrict__ A, long long strideAb,
                       const unsigned short* __restrict__ W,
                       const float* __restrict__ bih,
                       const float* __restrict__ bhh,
                       float* __restrict__ Gpre) {
    __shared__ unsigned short As[2][128 * 32];   // 16 KB
    __shared__ unsigned short Bs[2][64 * 32];    //  8 KB

    const int tid  = threadIdx.x;
    const int wave = tid >> 5;
    const int lane = tid & 31;
    const int m0 = blockIdx.x * 128;
    const int n0 = blockIdx.y * 64;
    const int z  = blockIdx.z;
    const unsigned short* Ab = A + (long long)z * strideAb;

    auto stage = [&](int buf, int kc) {
        // A tile: 128x32 bf16 = 512 x 16B, 2 per thread
        for (int i = tid; i < 512; i += RBLK) {
            int row = i >> 2, q = i & 3;
            async_cp16(lds_off(&As[buf][row * 32 + q * 8]),
                       Ab + (long long)(m0 + row) * HID + kc + q * 8);
        }
        // B tile (rows of W): 64x32 bf16 = 256 x 16B, 1 per thread
        {
            int row = tid >> 2, q = tid & 3;
            async_cp16(lds_off(&Bs[buf][row * 32 + q * 8]),
                       W + (long long)(n0 + row) * HID + kc + q * 8);
        }
    };

    v8f acc0 = {}, acc1 = {}, acc2 = {}, acc3 = {};

    const int m     = lane & 15;
    const int khalf = lane >> 4;
    const int bn = lane & 15;
    const int g  = lane >> 4;

    stage(0, 0);
    wait_async0();
    __syncthreads();

    int buf = 0;
    for (int kc = 0; kc < HID; kc += 32) {
        if (kc + 32 < HID) stage(buf ^ 1, kc + 32);   // prefetch next chunk

        // A fragment for this wave's 16-row subtile (contiguous k: k = 16*khalf + s)
        const unsigned short* ar = &As[buf][(wave * 16 + m) * 32];
        v16bf af = ld_frag(ar + khalf * 16, ar + khalf * 16 + 8);

        {
            const unsigned short* br = &Bs[buf][(0 * 16 + bn) * 32 + g * 16];
            acc0 = wmma_bf16(af, ld_frag(br, br + 8), acc0);
        }
        {
            const unsigned short* br = &Bs[buf][(1 * 16 + bn) * 32 + g * 16];
            acc1 = wmma_bf16(af, ld_frag(br, br + 8), acc1);
        }
        {
            const unsigned short* br = &Bs[buf][(2 * 16 + bn) * 32 + g * 16];
            acc2 = wmma_bf16(af, ld_frag(br, br + 8), acc2);
        }
        {
            const unsigned short* br = &Bs[buf][(3 * 16 + bn) * 32 + g * 16];
            acc3 = wmma_bf16(af, ld_frag(br, br + 8), acc3);
        }

        wait_async0();
        __syncthreads();
        buf ^= 1;
    }

    // epilogue: C layout 16x16 f32 — VGPR v: lanes0-15 (M=v), lanes16-31 (M=v+8)
    const int rowBase = m0 + wave * 16 + 8 * (lane >> 4);
    v8f accs[4] = {acc0, acc1, acc2, acc3};
#pragma unroll
    for (int ns = 0; ns < 4; ++ns) {
        int col = n0 + ns * 16 + (lane & 15);
        float bias = bih[col] + bhh[col];
#pragma unroll
        for (int v = 0; v < 8; ++v) {
            long long row = rowBase + v;
            Gpre[(long long)z * T_LEN * G4 + row * G4 + col] = accs[ns][v] + bias;
        }
    }
}

// ---------------------------------------------------------------------------
// Persistent recurrent kernel. 32 WGs, WG w owns hidden units [w*32, w*32+32)
// across all 4 gates; its 128x1024 W_hh slice lives in LDS (256 KB, bf16),
// streamed in once via async-to-LDS. Both branches batched as B columns 0/1.
// Per-step h exchange buffer is bf16 in global; staged with a single async
// 4 KB copy. Device-wide phase barrier via monotonic atomic counter.
// ---------------------------------------------------------------------------
__global__ void k_recurrent(const unsigned short* __restrict__ Whh,  // [4096][1024] bf16 (layer base)
                            const float* __restrict__ Gpre,          // [2][T][4096]
                            const float* __restrict__ h0a, const float* __restrict__ h0b,
                            const float* __restrict__ c0a, const float* __restrict__ c0b,
                            unsigned short* __restrict__ hbufb,      // [2][1024] bf16
                            unsigned short* __restrict__ seqOut,     // [2][T][1024] bf16 or null
                            float* __restrict__ finOut,              // [2][1024] or null
                            unsigned int* __restrict__ counter,
                            int writeSeq, int writeFinal) {
    extern __shared__ unsigned short smem[];
    unsigned short* sW = smem;                       // 128*1024 bf16 (256 KB)
    unsigned short* sH = smem + 128 * 1024;          // 2*1024 bf16
    float* sGate = (float*)(smem + 128 * 1024 + 2 * 1024); // [4][2][32]
    float* sC    = sGate + 4 * 2 * 32;               // [2][32]

    const int tid  = threadIdx.x;
    const int wave = tid >> 5;
    const int lane = tid & 31;
    const int wg   = blockIdx.x;

    // ---- async-stream this WG's W_hh slice into LDS ----
    for (int i = tid; i < 128 * 128; i += RBLK) {
        int r = i >> 7, q = i & 127;                 // row in slice, 16B chunk
        int grow = ((r >> 5) * HID) + wg * 32 + (r & 31);
        async_cp16(lds_off(&sW[r * HID + q * 8]),
                   Whh + (long long)grow * HID + q * 8);
    }
    wait_async0();
    __syncthreads();

    const int gate = wave >> 1;        // 0..3
    const int sub  = wave & 1;         // which 16-row half of the 32-unit slice
    const int m     = lane & 15;
    const int khalf = lane >> 4;
    const int arow  = gate * 32 + sub * 16 + m;
    const int bn = lane & 15;
    const int bg = lane >> 4;

    for (int t = 0; t < T_LEN; ++t) {
        // ---- stage h (both branches, bf16) into LDS ----
        if (t == 0) {
            for (int i = tid; i < 2 * HID; i += RBLK) {
                int br = i >> 10, j = i & (HID - 1);
                sH[i] = f2bf((br == 0) ? h0a[j] : h0b[j]);
            }
        } else {
            // 2*1024 bf16 = 4 KB = 256 x 16B async copies (one per thread)
            async_cp16(lds_off(&sH[tid * 8]), hbufb + tid * 8);
            wait_async0();
        }
        __syncthreads();

        // ---- WMMA: 16 gate rows x (h branch0, h branch1) ----
        v8f acc = {};
        const unsigned short* wr = &sW[arow * HID];
#pragma unroll 4
        for (int kc = 0; kc < HID; kc += 32) {
            v16bf af = ld_frag(wr + kc + khalf * 16, wr + kc + khalf * 16 + 8);
            FragU ub;
            if (bn < 2) {
                const unsigned short* hp = &sH[bn * HID + kc + bg * 16];
                ub.q[0] = *reinterpret_cast<const uint4*>(hp);
                ub.q[1] = *reinterpret_cast<const uint4*>(hp + 8);
            } else {
                ub.q[0] = make_uint4(0u, 0u, 0u, 0u);
                ub.q[1] = make_uint4(0u, 0u, 0u, 0u);
            }
            acc = wmma_bf16(af, ub.v, acc);
        }

        // ---- extract columns 0/1, add precomputed input projection ----
        if ((lane & 15) < 2) {
            int br = lane & 15;
            int ubase = sub * 16 + (lane >> 4) * 8;
            long long gpoff = (long long)br * T_LEN * G4 + (long long)t * G4
                            + gate * HID + wg * 32 + ubase;
#pragma unroll
            for (int v = 0; v < 8; ++v) {
                sGate[gate * 64 + br * 32 + ubase + v] = acc[v] + Gpre[gpoff + v];
            }
        }
        __syncthreads();

        // ---- LSTM cell update for this WG's 32 units x 2 branches ----
        if (tid < 64) {
            int br = tid >> 5, u = tid & 31;
            float gi = sigmoidf_(sGate[0 * 64 + br * 32 + u]);
            float gf = sigmoidf_(sGate[1 * 64 + br * 32 + u]);
            float gg = tanhf(sGate[2 * 64 + br * 32 + u]);
            float go = sigmoidf_(sGate[3 * 64 + br * 32 + u]);
            float c = (t == 0) ? ((br == 0) ? c0a[wg * 32 + u] : c0b[wg * 32 + u])
                               : sC[tid];
            c = gf * c + gi * gg;
            sC[tid] = c;
            float h = go * tanhf(c);
            int j = wg * 32 + u;
            unsigned short hb = f2bf(h);
            hbufb[br * HID + j] = hb;
            if (writeSeq)
                seqOut[(long long)br * T_LEN * HID + (long long)t * HID + j] = hb;
            if (writeFinal && t == T_LEN - 1)
                finOut[br * HID + j] = h;
        }

        // ---- device-wide phase barrier (monotonic counter) ----
        __threadfence();
        __syncthreads();
        if (tid == 0) {
            atomicAdd(counter, 1u);
            unsigned tgt = (unsigned)NWG * (unsigned)(t + 1);
            volatile unsigned int* vc = counter;
            while (*vc < tgt) { __builtin_amdgcn_s_sleep(8); }
        }
        __syncthreads();
        __threadfence();
    }
}

// ---------------------------------------------------------------------------
// Final: 5 * exp(-sum |o1 - o2|)
// ---------------------------------------------------------------------------
__global__ void k_final(const float* __restrict__ hfin, float* __restrict__ out) {
    __shared__ float red[256];
    int tid = threadIdx.x;
    float s = 0.0f;
    for (int i = tid; i < HID; i += 256)
        s += fabsf(hfin[i] - hfin[HID + i]);
    red[tid] = s;
    __syncthreads();
    for (int o = 128; o > 0; o >>= 1) {
        if (tid < o) red[tid] += red[tid + o];
        __syncthreads();
    }
    if (tid == 0) out[0] = 5.0f * __expf(-red[0]);
}

// ---------------------------------------------------------------------------
// launch
// ---------------------------------------------------------------------------
extern "C" void kernel_launch(void* const* d_in, const int* in_sizes, int n_in,
                              void* d_out, int out_size, void* d_ws, size_t ws_size,
                              hipStream_t stream) {
    const float* s1   = (const float*)d_in[0];
    const float* s2   = (const float*)d_in[1];
    const float* h1_0 = (const float*)d_in[2];
    const float* c1_0 = (const float*)d_in[3];
    const float* h2_0 = (const float*)d_in[4];
    const float* c2_0 = (const float*)d_in[5];
    const float* W_ih = (const float*)d_in[6];
    const float* W_hh = (const float*)d_in[7];
    const float* b_ih = (const float*)d_in[8];
    const float* b_hh = (const float*)d_in[9];
    float* out = (float*)d_out;

    // workspace layout
    char* w = (char*)d_ws;
    unsigned int* counters = (unsigned int*)w;                       // 256 B
    unsigned short* wih   = (unsigned short*)(w + 256);              // [2][4096][1024]
    unsigned short* whh   = wih   + (long long)2 * G4 * HID;
    unsigned short* sbf   = whh   + (long long)2 * G4 * HID;         // [2][T][1024]
    unsigned short* hseq  = sbf   + (long long)2 * T_LEN * HID;      // [2][T][1024]
    unsigned short* hbufb = hseq  + (long long)2 * T_LEN * HID;      // [2][1024] bf16
    float* gpre = (float*)(hbufb + 2 * HID);                         // [2][T][4096]
    float* hfin = gpre + (long long)2 * T_LEN * G4;                  // [2][1024]

    const size_t RSMEM = (size_t)128 * 1024 * 2 + 2 * 1024 * 2
                       + 4 * 2 * 32 * 4 + 2 * 32 * 4;                // 267520 B LDS

    k_init<<<1, 64, 0, stream>>>(counters);

    // fp32 -> bf16 conversions
    k_f2bf<<<2048, 256, 0, stream>>>(W_ih, wih, (long long)2 * G4 * HID);
    k_f2bf<<<2048, 256, 0, stream>>>(W_hh, whh, (long long)2 * G4 * HID);
    k_f2bf<<<2048, 256, 0, stream>>>(s1, sbf,                        (long long)T_LEN * HID);
    k_f2bf<<<2048, 256, 0, stream>>>(s2, sbf + (long long)T_LEN * HID, (long long)T_LEN * HID);

    dim3 ggrid(T_LEN / 128, G4 / 64, 2);

    // ---- layer 0 ----
    k_gemm<<<ggrid, RBLK, 0, stream>>>(sbf, (long long)T_LEN * HID,
                                       wih, b_ih, b_hh, gpre);
    k_recurrent<<<NWG, RBLK, RSMEM, stream>>>(whh, gpre,
                                              h1_0, h2_0, c1_0, c2_0,
                                              hbufb, hseq, nullptr,
                                              counters + 0, 1, 0);

    // ---- layer 1 ----
    k_gemm<<<ggrid, RBLK, 0, stream>>>(hseq, (long long)T_LEN * HID,
                                       wih + (long long)G4 * HID,
                                       b_ih + G4, b_hh + G4, gpre);
    k_recurrent<<<NWG, RBLK, RSMEM, stream>>>(whh + (long long)G4 * HID, gpre,
                                              h1_0 + HID, h2_0 + HID,
                                              c1_0 + HID, c2_0 + HID,
                                              hbufb, nullptr, hfin,
                                              counters + 1, 0, 1);

    k_final<<<1, 256, 0, stream>>>(hfin, out);
    (void)in_sizes; (void)n_in; (void)out_size; (void)ws_size;
}
